// ResGCN_63024350101699
// MI455X (gfx1250) — matrix operand order repaired and checked by reading, not compiled, not run on hardware
//
#include <hip/hip_runtime.h>
#include <hip/hip_bf16.h>

// ---------------------------------------------------------------------------
// ResGCN (ChebConv x16 + GraphNorm/residual) forward for MI455X (gfx1250).
// Dense GEMMs: v_wmma_f32_16x16x32_bf16, 16x64 tile per wave, pre-swizzled
// bf16 B-fragments. Sparse propagation: float4 gather + coalesced f32 atomics.
// ---------------------------------------------------------------------------

#define NNODES 50000
#define NEDGES 800000
#define MAXC   256
#define NEG_SLOPE 0.2f
#define GN_EPS 1e-5f

typedef __attribute__((ext_vector_type(16))) __bf16 v16bf;
typedef __attribute__((ext_vector_type(8)))  float  v8f;

// ------------------------------- utility -----------------------------------

__global__ void fill_zero_kernel(float* __restrict__ p, long n) {
    long t = (long)blockIdx.x * blockDim.x + threadIdx.x;
    if (t < n) p[t] = 0.0f;
}

// ------------------------- degree / edge norms ------------------------------

__global__ void deg_count_kernel(const int* __restrict__ src, float* __restrict__ deg, int ne) {
    int e = blockIdx.x * blockDim.x + threadIdx.x;
    if (e < ne) atomicAdd(&deg[src[e]], 1.0f);
}

__global__ void deg2dis_kernel(float* __restrict__ d, int n) {
    int i = blockIdx.x * blockDim.x + threadIdx.x;
    if (i < n) {
        float v = d[i];
        d[i] = (v > 0.0f) ? rsqrtf(v) : 0.0f;   // deg>0 -> rsqrt(max(deg,1)) == rsqrt(deg)
    }
}

__global__ void normw_kernel(const int* __restrict__ src, const int* __restrict__ dst,
                             const float* __restrict__ dis, float* __restrict__ nw, int ne) {
    int e = blockIdx.x * blockDim.x + threadIdx.x;
    if (e < ne) nw[e] = -dis[src[e]] * dis[dst[e]];
}

// ------------------------- sparse propagation -------------------------------
// out[dst, c] += norm_w[e] * h[src, c]  (thread per (edge, 4 channels))
__global__ void prop_kernel(const float* __restrict__ h, const int* __restrict__ src,
                            const int* __restrict__ dst, const float* __restrict__ nw,
                            float* __restrict__ out, int Cq, long totalq) {
    long tid = (long)blockIdx.x * blockDim.x + threadIdx.x;
    if (tid >= totalq) return;
    int  cq = (int)(tid % Cq);
    long e  = tid / Cq;
    int  s  = src[e];
    int  d  = dst[e];
    float w = nw[e];
    const float4 hv = *(const float4*)(h + ((long)s * Cq + cq) * 4);
    float* op = out + ((long)d * Cq + cq) * 4;
    atomicAdd(op + 0, w * hv.x);
    atomicAdd(op + 1, w * hv.y);
    atomicAdd(op + 2, w * hv.z);
    atomicAdd(op + 3, w * hv.w);
}

// Tx2 = 2*prop(Tx1) - Tx0  (p already holds prop(Tx1))
__global__ void tx2_combine_kernel(float* __restrict__ p, const float* __restrict__ x0, long n) {
    long t = (long)blockIdx.x * blockDim.x + threadIdx.x;
    if (t < n) p[t] = 2.0f * p[t] - x0[t];
}

// --------------------- weight pre-swizzle (f32 -> bf16 B-frag) --------------
// B-fragment layout (16x16x32 bf16 WMMA): for tile (colT, kc):
//   lane 0-15  : col = colT*16 + lane,      K = kc*32 + e      (e = 0..15)
//   lane 16-31 : col = colT*16 + lane-16,   K = kc*32 + 16 + e
// Stored as out[(((colT*kcN + kc)*32 + lane)*16 + e)]  (contiguous per lane).
__global__ void swizzle_w_kernel(const float* __restrict__ W, unsigned short* __restrict__ out,
                                 int fi, int fo) {
    int t = blockIdx.x * blockDim.x + threadIdx.x;
    if (t >= fi * fo) return;
    int e    = t & 15;
    int lane = (t >> 4) & 31;
    int rest = t >> 9;
    int kcN  = fi >> 5;
    int kc   = rest % kcN;
    int colT = rest / kcN;
    int k    = kc * 32 + (lane >> 4) * 16 + e;
    int col  = colT * 16 + (lane & 15);
    __bf16 b = (__bf16)W[(long)k * fo + col];
    out[t] = __builtin_bit_cast(unsigned short, b);
}

// ------------------------------ WMMA GEMM -----------------------------------
// out = A0@W0 + A1@W1 (+ A2@W2) + bias.  A: N x fi f32 row-major; B pre-
// swizzled bf16 fragments. One wave computes a 16x64 output tile (4 WMMAs
// per k-step sharing a single A fragment).

__device__ __forceinline__ void wmma_term4(const float* __restrict__ A,
                                           const unsigned short* __restrict__ B,
                                           v8f acc[4], int row, int lane,
                                           int half, int colBase, int fi) {
    const int kcN = fi >> 5;
    for (int kc = 0; kc < kcN; ++kc) {
        // ---- A fragment: VGPRs 0..3: K = kc*32 + half*16 + {0..7}
        //                  VGPRs 4..7: K = kc*32 + half*16 + 8 + {0..7}
        const float* ap = A + (long)row * fi + kc * 32 + half * 16;
        float4 a0 = *(const float4*)(ap);
        float4 a1 = *(const float4*)(ap + 4);
        float4 a2 = *(const float4*)(ap + 8);
        float4 a3 = *(const float4*)(ap + 12);
        v16bf af;
        af[0]  = (__bf16)a0.x; af[1]  = (__bf16)a0.y; af[2]  = (__bf16)a0.z; af[3]  = (__bf16)a0.w;
        af[4]  = (__bf16)a1.x; af[5]  = (__bf16)a1.y; af[6]  = (__bf16)a1.z; af[7]  = (__bf16)a1.w;
        af[8]  = (__bf16)a2.x; af[9]  = (__bf16)a2.y; af[10] = (__bf16)a2.z; af[11] = (__bf16)a2.w;
        af[12] = (__bf16)a3.x; af[13] = (__bf16)a3.y; af[14] = (__bf16)a3.z; af[15] = (__bf16)a3.w;
#pragma unroll
        for (int j = 0; j < 4; ++j) {
            const unsigned short* bp =
                B + ((((long)(colBase + j) * kcN + kc) * 32 + lane) << 4);
            union { uint4 u[2]; v16bf v; } ub;
            ub.u[0] = ((const uint4*)bp)[0];
            ub.u[1] = ((const uint4*)bp)[1];
            acc[j] = __builtin_amdgcn_wmma_f32_16x16x32_bf16(
                false, af, false, ub.v, (short)0, acc[j], false, false);
        }
    }
}

__global__ void __launch_bounds__(256)
cheb_matmul_kernel(const float* __restrict__ A0, const float* __restrict__ A1,
                   const float* __restrict__ A2,
                   const unsigned short* __restrict__ B0,
                   const unsigned short* __restrict__ B1,
                   const unsigned short* __restrict__ B2,
                   const float* __restrict__ bias, float* __restrict__ out,
                   int nterms, int fi, int fo) {
    const int lane    = threadIdx.x & 31;
    const int wave    = threadIdx.x >> 5;
    const int rowTile = blockIdx.x * 8 + wave;
    if (rowTile * 16 >= NNODES) return;            // wave-uniform: EXEC stays all-1s
    const int colBase = blockIdx.y * 4;            // 4 x 16-col tiles = 64 cols
    const int half    = lane >> 4;
    const int l15     = lane & 15;
    const int row     = rowTile * 16 + l15;

    v8f acc[4] = {{}, {}, {}, {}};
    wmma_term4(A0, B0, acc, row, lane, half, colBase, fi);
    wmma_term4(A1, B1, acc, row, lane, half, colBase, fi);
    if (nterms > 2) wmma_term4(A2, B2, acc, row, lane, half, colBase, fi);

    // C/D layout: VGPR r -> M = r + 8*half, N = l15
#pragma unroll
    for (int j = 0; j < 4; ++j) {
        int col = (colBase + j) * 16 + l15;
        float b = bias[col];
#pragma unroll
        for (int r = 0; r < 8; ++r) {
            int orow = rowTile * 16 + r + half * 8;
            out[(long)orow * fo + col] = acc[j][r] + b;
        }
    }
}

// ------------------------------ GraphNorm -----------------------------------

__global__ void colstats_kernel(const float* __restrict__ h, float* __restrict__ sum,
                                float* __restrict__ sumsq, int C) {
    int c = threadIdx.x % C;
    int g = threadIdx.x / C;
    int G = blockDim.x / C;
    long r0   = (long)blockIdx.x * 64;
    long rend = r0 + 64; if (rend > NNODES) rend = NNODES;
    float s = 0.0f, s2 = 0.0f;
    for (long r = r0 + g; r < rend; r += G) {
        float v = h[r * C + c];
        s += v; s2 += v * v;
    }
    atomicAdd(&sum[c], s);
    if (sumsq) atomicAdd(&sumsq[c], s2);
}

__global__ void finstats_kernel(const float* __restrict__ sum, const float* __restrict__ sumsq,
                                const float* __restrict__ alpha, float* __restrict__ meanAM,
                                float* __restrict__ inv, int C) {
    int c = blockIdx.x * blockDim.x + threadIdx.x;
    if (c >= C) return;
    const float invN = 1.0f / (float)NNODES;
    float m = sum[c] * invN;
    float a = alpha[c];
    // E[(h - a*m)^2] = E[h^2] - 2*a*m^2 + a^2*m^2
    float var = sumsq[c] * invN - 2.0f * a * m * m + a * a * m * m;
    meanAM[c] = a * m;
    inv[c]    = rsqrtf(var + GN_EPS);
}

__global__ void norm_act_kernel(float* __restrict__ h, const float* __restrict__ gw,
                                const float* __restrict__ gb, const float* __restrict__ meanAM,
                                const float* __restrict__ inv, int C, long n) {
    long t = (long)blockIdx.x * blockDim.x + threadIdx.x;
    if (t >= n) return;
    int c = (int)(t % C);
    float v = (h[t] - meanAM[c]) * inv[c] * gw[c] + gb[c];
    h[t] = (v > 0.0f) ? v : NEG_SLOPE * v;
}

__global__ void residual_relu_kernel(float* __restrict__ h, const float* __restrict__ x, long n) {
    long t = (long)blockIdx.x * blockDim.x + threadIdx.x;
    if (t >= n) return;
    float v = h[t] + x[t];
    h[t] = (v > 0.0f) ? v : 0.0f;
}

// ------------------------------- head ---------------------------------------

__global__ void final_lin_kernel(const float* __restrict__ colsum, const float* __restrict__ linW,
                                 const float* __restrict__ linB, float* __restrict__ out) {
    int o = threadIdx.x;
    if (o >= 64) return;
    const float invN = 1.0f / (float)NNODES;
    float acc = 0.0f;
    for (int c = 0; c < 64; ++c)
        acc += (colsum[c] * invN) * linW[c * 64 + o];
    out[o] = acc + linB[o];
}

// ---------------------------------------------------------------------------
//                               host launcher
// ---------------------------------------------------------------------------

extern "C" void kernel_launch(void* const* d_in, const int* in_sizes, int n_in,
                              void* d_out, int out_size, void* d_ws, size_t ws_size,
                              hipStream_t stream) {
    (void)in_sizes; (void)n_in; (void)out_size; (void)ws_size;
    const int N = NNODES, E = NEDGES;

    // ---- inputs (jax pytree flatten order; None entries dropped) ----
    int idx = 0;
    const int*   edge = (const int*)d_in[idx++];          // [2,E]
    const int*   src  = edge;
    const int*   dst  = edge + E;
    const float* x    = (const float*)d_in[idx++];        // [N,64]

    static const int ksPat[4] = {2, 3, 3, 2};
    static const int fiPat[4] = {64, 128, 256, 128};
    static const int foPat[4] = {128, 256, 128, 64};
    const float* W[16][3];
    const float* Bias[16];
    int Ks[16], Fi[16], Fo[16];
    for (int i = 0; i < 16; ++i) {
        int m = i % 4;
        Ks[i] = ksPat[m]; Fi[i] = fiPat[m]; Fo[i] = foPat[m];
        for (int k = 0; k < Ks[i]; ++k) W[i][k] = (const float*)d_in[idx++];
        if (Ks[i] < 3) W[i][2] = nullptr;
        Bias[i] = (const float*)d_in[idx++];
    }
    const float *Gw[16], *Gb[16], *Ga[16];
    for (int i = 0; i < 16; ++i) {
        if (i % 4 != 3) {
            Gw[i] = (const float*)d_in[idx++];
            Gb[i] = (const float*)d_in[idx++];
            Ga[i] = (const float*)d_in[idx++];
        } else { Gw[i] = Gb[i] = Ga[i] = nullptr; }
    }
    const float* linW = (const float*)d_in[idx++];
    const float* linB = (const float*)d_in[idx++];

    // ---- workspace carve-up ----
    char* p = (char*)d_ws;
    auto alloc = [&](size_t bytes) -> void* {
        void* r = (void*)p;
        p += (bytes + 255) & ~(size_t)255;
        return r;
    };
    float* dis  = (float*)alloc((size_t)N * sizeof(float));
    float* nw   = (float*)alloc((size_t)E * sizeof(float));
    float* bufs[4];
    for (int i = 0; i < 4; ++i) bufs[i] = (float*)alloc((size_t)N * MAXC * sizeof(float));
    float* s_sum   = (float*)alloc(MAXC * sizeof(float));
    float* s_sumsq = (float*)alloc(MAXC * sizeof(float));
    float* s_mean  = (float*)alloc(MAXC * sizeof(float));
    float* s_inv   = (float*)alloc(MAXC * sizeof(float));

    // bf16 swizzled weight fragments for all 40 conv weights
    unsigned short* Bsw[16][3];
    {
        size_t total = 0;
        for (int i = 0; i < 16; ++i) total += (size_t)Ks[i] * Fi[i] * Fo[i];
        unsigned short* base = (unsigned short*)alloc(total * sizeof(unsigned short));
        size_t cur = 0;
        for (int i = 0; i < 16; ++i) {
            for (int k = 0; k < 3; ++k) Bsw[i][k] = nullptr;
            for (int k = 0; k < Ks[i]; ++k) {
                Bsw[i][k] = base + cur;
                cur += (size_t)Fi[i] * Fo[i];
            }
        }
    }
    for (int i = 0; i < 16; ++i)
        for (int k = 0; k < Ks[i]; ++k) {
            int total = Fi[i] * Fo[i];
            swizzle_w_kernel<<<(total + 255) / 256, 256, 0, stream>>>(
                W[i][k], Bsw[i][k], Fi[i], Fo[i]);
        }

    bool used[4] = {false, false, false, false};
    auto grab = [&]() -> int {
        for (int i = 0; i < 4; ++i) if (!used[i]) { used[i] = true; return i; }
        return 0; // unreachable by construction
    };

    auto launch_fill = [&](float* ptr, long n) {
        int blocks = (int)((n + 255) / 256);
        fill_zero_kernel<<<blocks, 256, 0, stream>>>(ptr, n);
    };

    // ---- edge normalization: deg -> dis -> norm_w ----
    launch_fill(dis, N);
    deg_count_kernel<<<(E + 255) / 256, 256, 0, stream>>>(src, dis, E);
    deg2dis_kernel<<<(N + 255) / 256, 256, 0, stream>>>(dis, N);
    normw_kernel<<<(E + 255) / 256, 256, 0, stream>>>(src, dst, dis, nw, E);

    // ---- 16 ChebConv layers ----
    const float* h = x;     // Tx0 of layer 0 is the input itself
    int hOwned = -1;
    for (int i = 0; i < 16; ++i) {
        const int Ci = Fi[i], Co = Fo[i], K = Ks[i];
        const long nIn  = (long)N * Ci;
        const long nOut = (long)N * Co;
        const int  Cq   = Ci / 4;
        const long eTot = (long)E * Cq;
        const int  eBlocks = (int)((eTot + 255) / 256);

        // Tx1 = prop(h)
        int iT1 = grab();
        float* T1 = bufs[iT1];
        launch_fill(T1, nIn);
        prop_kernel<<<eBlocks, 256, 0, stream>>>(h, src, dst, nw, T1, Cq, eTot);

        // Tx2 = 2*prop(Tx1) - Tx0
        int iT2 = -1; float* T2 = nullptr;
        if (K == 3) {
            iT2 = grab(); T2 = bufs[iT2];
            launch_fill(T2, nIn);
            prop_kernel<<<eBlocks, 256, 0, stream>>>(T1, src, dst, nw, T2, Cq, eTot);
            tx2_combine_kernel<<<(int)((nIn + 255) / 256), 256, 0, stream>>>(T2, h, nIn);
        }

        // out = Tx0@W0 + Tx1@W1 (+ Tx2@W2) + b   (WMMA bf16 -> f32)
        int iOut = grab();
        float* out = bufs[iOut];
        dim3 grid((N / 16 + 7) / 8, Co / 64);
        cheb_matmul_kernel<<<grid, 256, 0, stream>>>(h, T1, T2,
                                                     Bsw[i][0], Bsw[i][1], Bsw[i][2],
                                                     Bias[i], out, K, Ci, Co);
        used[iT1] = false;
        if (iT2 >= 0) used[iT2] = false;

        if (i % 4 != 3) {
            // GraphNorm + LeakyReLU
            launch_fill(s_sum, Co);
            launch_fill(s_sumsq, Co);
            colstats_kernel<<<(N + 63) / 64, 256, 0, stream>>>(out, s_sum, s_sumsq, Co);
            finstats_kernel<<<1, Co, 0, stream>>>(s_sum, s_sumsq, Ga[i], s_mean, s_inv, Co);
            norm_act_kernel<<<(int)((nOut + 255) / 256), 256, 0, stream>>>(
                out, Gw[i], Gb[i], s_mean, s_inv, Co, nOut);
        } else {
            // h = relu(out + x)   (Co == 64 here)
            residual_relu_kernel<<<(int)((nOut + 255) / 256), 256, 0, stream>>>(out, x, nOut);
        }

        if (hOwned >= 0) used[hOwned] = false;
        h = out; hOwned = iOut;
    }

    // ---- global mean pool + linear head ----
    launch_fill(s_sum, 64);
    colstats_kernel<<<(N + 63) / 64, 256, 0, stream>>>(h, s_sum, nullptr, 64);
    final_lin_kernel<<<1, 64, 0, stream>>>(s_sum, linW, linB, (float*)d_out);
}
